// LocalAttention_20942260535692
// MI455X (gfx1250) — matrix operand order repaired and checked by reading, not compiled, not run on hardware
//
#include <hip/hip_runtime.h>

#define B_ 2
#define S_ 8192
#define H_ 16
#define D_ 64
#define W_ 256
#define NB_ (S_ / W_)          // 32 query blocks
#define SCALE_ 0.125f          // 64^-0.5
#define NEG_INF_ -1e30f

typedef __attribute__((ext_vector_type(16))) _Float16 v16h;
typedef __attribute__((ext_vector_type(8)))  float    v8f;
typedef __attribute__((ext_vector_type(4)))  float    v4f;

// Load 8+8 contiguous f32 (two runs) -> one v16h A-fragment slice
__device__ __forceinline__ v16h load_frag_8_8(const float* __restrict__ p, int off0, int off1) {
    v4f a0 = *(const v4f*)(p + off0);
    v4f a1 = *(const v4f*)(p + off0 + 4);
    v4f b0 = *(const v4f*)(p + off1);
    v4f b1 = *(const v4f*)(p + off1 + 4);
    v16h r;
    r[0]=(_Float16)a0.x; r[1]=(_Float16)a0.y; r[2]=(_Float16)a0.z; r[3]=(_Float16)a0.w;
    r[4]=(_Float16)a1.x; r[5]=(_Float16)a1.y; r[6]=(_Float16)a1.z; r[7]=(_Float16)a1.w;
    r[8]=(_Float16)b0.x; r[9]=(_Float16)b0.y; r[10]=(_Float16)b0.z; r[11]=(_Float16)b0.w;
    r[12]=(_Float16)b1.x; r[13]=(_Float16)b1.y; r[14]=(_Float16)b1.z; r[15]=(_Float16)b1.w;
    return r;
}

// Load 16 contiguous f32 with scale -> v16h B-fragment slice (Q^T)
__device__ __forceinline__ v16h load_frag_16s(const float* __restrict__ p, float scale) {
    v16h r;
#pragma unroll
    for (int j = 0; j < 16; j += 4) {
        v4f x = *(const v4f*)(p + j);
        r[j+0]=(_Float16)(x.x*scale); r[j+1]=(_Float16)(x.y*scale);
        r[j+2]=(_Float16)(x.z*scale); r[j+3]=(_Float16)(x.w*scale);
    }
    return r;
}

__global__ void __launch_bounds__(512, 1)
local_attn_wmma(const float* __restrict__ Q, const float* __restrict__ K,
                const float* __restrict__ V, float* __restrict__ Out) {
    // V staged transposed as f16: VT[d][key] for one 256-key chunk (32 KB)
    __shared__ __align__(32) _Float16 VT[D_ * W_];

    const int n = blockIdx.x;       // query block
    const int h = blockIdx.y;
    const int b = blockIdx.z;

    const int tid  = threadIdx.x;
    const int lane = tid & 31;
    const int w    = tid >> 5;      // wave 0..15: query tile
    const int lh   = lane & 15;
    const int hi   = lane >> 4;     // half-wave select

    const size_t rowstride = (size_t)H_ * D_;
    const size_t bh = ((size_t)b * S_) * rowstride + (size_t)h * D_;

    // ---- Q^T B-fragments (resident): B elem (k=d, n=query) = Q[q][d], scale folded
    const int qloc = w * 16 + lh;                       // query within block (per lane)
    v16h qB[2];
    {
        const float* qp = Q + bh + (size_t)(n * W_ + qloc) * rowstride;
#pragma unroll
        for (int s = 0; s < 2; ++s)
            qB[s] = load_frag_16s(qp + 32 * s + 16 * hi, SCALE_);
    }

    // ---- accumulators: O is 16x64 as 4 WMMA C tiles; per-lane softmax stats
    v8f Oacc[4];
#pragma unroll
    for (int s = 0; s < 4; ++s)
#pragma unroll
        for (int r = 0; r < 8; ++r) Oacc[s][r] = 0.f;
    float mrun = NEG_INF_, lrun = 0.f;

    const int cstart = (n == 0) ? 1 : 0;   // block 0 has no previous block
    for (int c = cstart; c < 2; ++c) {
        __syncthreads();
        // ---- stage V chunk transposed into LDS (keys rbase..rbase+255)
        const int rbase = (n - 1) * W_ + c * W_;
#pragma unroll
        for (int i = 0; i < 16; ++i) {
            int p  = tid + i * 512;            // 8192 key-pair x d elements
            int dd = p & 63;                   // d
            int kp = p >> 6;                   // key pair 0..127
            const float* vp = V + bh + (size_t)(rbase + 2 * kp) * rowstride + dd;
            float v0 = vp[0];
            float v1 = vp[rowstride];
            VT[dd * W_ + 2 * kp]     = (_Float16)v0;
            VT[dd * W_ + 2 * kp + 1] = (_Float16)v1;
        }
        __syncthreads();

#pragma unroll
        for (int it = 0; it < 8; ++it) {
            const int kb = it * 32;            // key offset within chunk
            // ---- K A-fragments: 2 key tiles x 2 d-slabs (A = K_tile 16x32)
            v16h kA[2][2];
#pragma unroll
            for (int t = 0; t < 2; ++t) {
                const float* kp2 = K + bh + (size_t)(rbase + kb + t * 16 + lh) * rowstride;
#pragma unroll
                for (int s = 0; s < 2; ++s)
                    kA[t][s] = load_frag_8_8(kp2, 32 * s + 16 * hi, 32 * s + 16 * hi + 8);
            }
            // ---- S^T tiles: (16 keys x 16 queries) = K_tile x Q^T
            v8f c0, c1;
#pragma unroll
            for (int r = 0; r < 8; ++r) { c0[r] = 0.f; c1[r] = 0.f; }
            c0 = __builtin_amdgcn_wmma_f32_16x16x32_f16(false, kA[0][0], false, qB[0], (short)0, c0, false, false);
            c0 = __builtin_amdgcn_wmma_f32_16x16x32_f16(false, kA[0][1], false, qB[1], (short)0, c0, false, false);
            c1 = __builtin_amdgcn_wmma_f32_16x16x32_f16(false, kA[1][0], false, qB[0], (short)0, c1, false, false);
            c1 = __builtin_amdgcn_wmma_f32_16x16x32_f16(false, kA[1][1], false, qB[1], (short)0, c1, false, false);

            // ---- band mask: valid iff qloc < k2 <= qloc + W  (k2 in [prev|cur] coords)
#pragma unroll
            for (int r = 0; r < 8; ++r) {
                int k2a = c * W_ + kb + (r + 8 * hi);
                int k2b = k2a + 16;
                c0[r] = (k2a > qloc && k2a <= qloc + W_) ? c0[r] : NEG_INF_;
                c1[r] = (k2b > qloc && k2b <= qloc + W_) ? c1[r] : NEG_INF_;
            }

            // ---- online softmax (per-query stats: query = lane%16)
            float tm = NEG_INF_;
#pragma unroll
            for (int r = 0; r < 8; ++r) tm = fmaxf(tm, fmaxf(c0[r], c1[r]));
            tm = fmaxf(tm, __shfl_xor(tm, 16));
            float nm   = fmaxf(mrun, tm);
            float corr = __expf(mrun - nm);
            mrun = nm;

            float ps = 0.f;
            v16h pA;                            // P^T lands directly in A-layout
#pragma unroll
            for (int r = 0; r < 8; ++r) {
                float p0 = __expf(c0[r] - nm);
                float p1 = __expf(c1[r] - nm);
                ps += p0 + p1;
                pA[r]     = (_Float16)p0;
                pA[8 + r] = (_Float16)p1;
            }
            ps += __shfl_xor(ps, 16);
            lrun = lrun * corr + ps;

            // ---- rescale O: query of O-VGPR r is (r + 8*hi); stat lives at lane q%16
            float f[8];
#pragma unroll
            for (int r = 0; r < 8; ++r) f[r] = __shfl(corr, r + 8 * hi);
#pragma unroll
            for (int s = 0; s < 4; ++s)
#pragma unroll
                for (int r = 0; r < 8; ++r) Oacc[s][r] *= f[r];

            // ---- O += P x V  (B-frag from transposed LDS: keys 8*hi+j and 16+8*hi+j per lane)
#pragma unroll
            for (int s = 0; s < 4; ++s) {
                const _Float16* vrow = VT + (16 * s + lh) * W_ + kb;
                v16h vb;
#pragma unroll
                for (int j = 0; j < 8; ++j) { vb[j] = vrow[8 * hi + j]; vb[8 + j] = vrow[16 + 8 * hi + j]; }
                Oacc[s] = __builtin_amdgcn_wmma_f32_16x16x32_f16(false, pA, false, vb, (short)0, Oacc[s], false, false);
            }
        }
    }

    // ---- normalize and store (f32 out, coalesced across lanes)
    float inv[8];
#pragma unroll
    for (int r = 0; r < 8; ++r) inv[r] = 1.f / __shfl(lrun, r + 8 * hi);
    const int qrow0 = n * W_ + w * 16;
#pragma unroll
    for (int s = 0; s < 4; ++s)
#pragma unroll
        for (int r = 0; r < 8; ++r) {
            int qrow = qrow0 + r + 8 * hi;
            int d    = 16 * s + lh;
            Out[bh + (size_t)qrow * rowstride + d] = Oacc[s][r] * inv[r];
        }
}

extern "C" void kernel_launch(void* const* d_in, const int* in_sizes, int n_in,
                              void* d_out, int out_size, void* d_ws, size_t ws_size,
                              hipStream_t stream) {
    (void)in_sizes; (void)n_in; (void)d_ws; (void)ws_size; (void)out_size;
    const float* q = (const float*)d_in[0];
    const float* k = (const float*)d_in[1];
    const float* v = (const float*)d_in[2];
    float* out = (float*)d_out;
    dim3 grid(NB_, H_, B_);     // 32 x 16 x 2 workgroups
    dim3 block(512);            // 16 wave32
    local_attn_wmma<<<grid, block, 0, stream>>>(q, k, v, out);
}
